// ComplexAttention_14929306321448
// MI455X (gfx1250) — matrix operand order repaired and checked by reading, not compiled, not run on hardware
//
#include <hip/hip_runtime.h>
#include <hip/hip_bf16.h>

// ---------------------------------------------------------------------------
// Complex attention for MI455X (gfx1250, wave32).
// GEMMs: v_wmma_f32_16x16x32_bf16, tiles staged by the Tensor Data Mover
// (tensor_load_to_lds + s_wait_tensorcnt), double-buffered LDS.
// ---------------------------------------------------------------------------

typedef __bf16 bf16_t;
typedef __attribute__((ext_vector_type(16))) __bf16 v16bf;
typedef __attribute__((ext_vector_type(8)))  float  v8f;
typedef __attribute__((ext_vector_type(4)))  unsigned int v4u;
typedef __attribute__((ext_vector_type(4)))  int v4i;
typedef __attribute__((ext_vector_type(8)))  int v8i;

#define B_    4
#define S_    2048
#define DIM_  1024
#define MTOT  (B_ * S_)      // 8192 projection rows

#define BM 128               // block tile M
#define BN 64                // block tile N
#define TK 32                // K step (16x16x32 bf16 WMMA)
#define LDA 40               // LDS A pitch = TK + 8  (TDM pad: 16 dw + 4 dw)
#define LDB 72               // LDS B pitch = BN + 8  (TDM pad: 32 dw + 4 dw)

union FragU { uint4 u[2]; v16bf v; };

static __device__ __forceinline__ v8f zero8() {
  v8f z;
#pragma unroll
  for (int e = 0; e < 8; ++e) z[e] = 0.0f;
  return z;
}

static __device__ __forceinline__ v8f wmma_bf16(const uint4& alo, const uint4& ahi,
                                                const uint4& blo, const uint4& bhi,
                                                v8f c) {
  FragU a, b;
  a.u[0] = alo; a.u[1] = ahi;
  b.u[0] = blo; b.u[1] = bhi;
  return __builtin_amdgcn_wmma_f32_16x16x32_bf16(false, a.v, false, b.v,
                                                 (short)0, c, false, false);
}

// A-fragment (16x32 bf16, MxK): lane = M row; lanes 0-15 K halves {0,16},
// lanes 16-31 K halves {8,24}. Two ds_load_b128 per fragment.
static __device__ __forceinline__ void load_a_frag(const bf16_t* sA, int mt, int lane,
                                                   uint4& lo, uint4& hi) {
  const int m  = mt + (lane & 15);
  const int kh = (lane >> 4) << 3;               // 0 or 8
  lo = *(const uint4*)(sA + m * LDA + kh);
  hi = *(const uint4*)(sA + m * LDA + 16 + kh);
}

// B-fragment (32x16 bf16, KxN): lane = N column; lanes 0-15 K halves {0,16},
// lanes 16-31 K halves {8,24} (matches A-fragment K order).
static __device__ __forceinline__ void load_b_frag(const bf16_t* sB, int nt, int lane,
                                                   uint4& lo, uint4& hi) {
  const int n  = nt + (lane & 15);
  const int kh = (lane >> 4) << 3;
  FragU f;
#pragma unroll
  for (int s = 0; s < 8; ++s) {
    f.v[s]     = sB[(kh + s) * LDB + n];
    f.v[8 + s] = sB[(16 + kh + s) * LDB + n];
  }
  lo = f.u[0];
  hi = f.u[1];
}

// ---------------------------------------------------------------------------
// Tensor Data Mover: async 2D tile load Global -> LDS (ISA 08, D# groups 0/1).
//   tile_d0 elements per line, tile_d1 lines, line stride stride0 (elements),
//   bf16 data (data_size code 1). LDS padding reproduces LDA/LDB pitches:
//   pad_int_code: pad after 2^(code+1) DWORDs; pad_amt_code: (code+1) DWORDs.
// ---------------------------------------------------------------------------
static __device__ __forceinline__ void tdm_load_2d(unsigned lds_byte_addr,
                                                   const bf16_t* gptr,
                                                   unsigned tile_d0, unsigned tile_d1,
                                                   unsigned stride0_elems,
                                                   unsigned pad_int_code,
                                                   unsigned pad_amt_code) {
  const unsigned long long ga = (unsigned long long)(uintptr_t)gptr;
  v4u g0;
  g0.x = 1u;                                               // count=1, user mode
  g0.y = lds_byte_addr;                                    // LDS dest (bytes)
  g0.z = (unsigned)(ga & 0xffffffffull);                   // global_addr[31:0]
  g0.w = (unsigned)((ga >> 32) & 0x1ffffffull) | (2u << 30);  // [56:32] | type=2
  v8i g1;
  g1[0] = (int)((1u << 16)                                 // data_size = 2 bytes
                | (1u << 20)                               // pad_enable
                | (pad_int_code << 22) | (pad_amt_code << 25));
  g1[1] = (int)((tile_d0 & 0xffffu) << 16);                // tensor_dim0 lo16
  g1[2] = (int)(((tile_d0 >> 16) & 0xffffu)                // tensor_dim0 hi16
                | ((tile_d1 & 0xffffu) << 16));            // tensor_dim1 lo16
  g1[3] = (int)(((tile_d1 >> 16) & 0xffffu)                // tensor_dim1 hi16
                | ((tile_d0 & 0xffffu) << 16));            // tile_dim0
  g1[4] = (int)(tile_d1 & 0xffffu);                        // tile_dim1, tile_dim2=0
  g1[5] = (int)stride0_elems;                              // dim0 stride [31:0]
  g1[6] = 0;
  g1[7] = 0;
  const v4i z4 = {0, 0, 0, 0};
#if __clang_major__ >= 23
  const v8i z8 = {0, 0, 0, 0, 0, 0, 0, 0};
  __builtin_amdgcn_tensor_load_to_lds(g0, g1, z4, z4, z8, 0);
#else
  __builtin_amdgcn_tensor_load_to_lds(g0, g1, z4, z4, 0);
#endif
}

static __device__ __forceinline__ unsigned lds_addr_of(const void* p) {
  return (unsigned)(uintptr_t)p;   // low 32 bits of LDS aperture = byte offset
}

// ---------------------------------------------------------------------------
// Pass 0a: fp32 -> bf16
// ---------------------------------------------------------------------------
__global__ __launch_bounds__(256) void convert_kernel(const float* __restrict__ x,
                                                      bf16_t* __restrict__ y, int n) {
  const int i = (blockIdx.x * 256 + threadIdx.x) * 4;
  if (i + 3 < n) {
    float4 v = *(const float4*)(x + i);
    y[i + 0] = (bf16_t)v.x;
    y[i + 1] = (bf16_t)v.y;
    y[i + 2] = (bf16_t)v.z;
    y[i + 3] = (bf16_t)v.w;
  }
}

// ---------------------------------------------------------------------------
// Pass 0b: weight transpose + convert: wt[k*D+n] = w[n*D+k]
// ---------------------------------------------------------------------------
__global__ __launch_bounds__(256) void wtrans_kernel(const float* __restrict__ w,
                                                     bf16_t* __restrict__ wt) {
  const int idx = blockIdx.x * 256 + threadIdx.x;  // k*DIM_ + n
  const int k = idx >> 10;
  const int n = idx & (DIM_ - 1);
  wt[idx] = (bf16_t)w[(size_t)n * DIM_ + k];
}

// ---------------------------------------------------------------------------
// Pass 0c: per-batch transpose of K projections: out[b][d][t] = in[b][t][d]
// (so the scores kernel's B tiles are plain 2D TDM loads)
// ---------------------------------------------------------------------------
__global__ __launch_bounds__(256) void ktrans_kernel(const bf16_t* __restrict__ in,
                                                     bf16_t* __restrict__ out) {
  const size_t idx = (size_t)blockIdx.x * 256 + threadIdx.x;
  const int t = (int)(idx & (S_ - 1));
  const int d = (int)((idx >> 11) & (DIM_ - 1));
  const int b = (int)(idx >> 21);
  out[idx] = in[((size_t)(b * S_ + t)) * DIM_ + d];
}

// ---------------------------------------------------------------------------
// Pass 1: complex projection.
//   P1=xr*Wr^T P2=xi*Wi^T P3=xi*Wr^T P4=xr*Wi^T ; yr=(P1-P2)s, yi=(P3+P4)s
// ---------------------------------------------------------------------------
__global__ __launch_bounds__(256) void proj_kernel(
    const bf16_t* __restrict__ xr, const bf16_t* __restrict__ xi,
    const bf16_t* __restrict__ wrt, const bf16_t* __restrict__ wit,
    bf16_t* __restrict__ yr, bf16_t* __restrict__ yi, float scale) {
  const int nb   = blockIdx.x * BN;
  const int mb   = blockIdx.y * BM;
  const int tid  = threadIdx.x;
  const int lane = tid & 31;
  const int wave = tid >> 5;
  const int wm = (wave >> 1) << 5;
  const int wn = (wave & 1) << 5;
  const bool issuer = (__builtin_amdgcn_readfirstlane((int)threadIdx.x) == 0);

  __shared__ __align__(16) bf16_t sAr[2][BM * LDA];
  __shared__ __align__(16) bf16_t sAi[2][BM * LDA];
  __shared__ __align__(16) bf16_t sBr[2][TK * LDB];
  __shared__ __align__(16) bf16_t sBi[2][TK * LDB];

  auto issue = [&](int kb, int buf) {
    const size_t ka = (size_t)kb * TK;
    tdm_load_2d(lds_addr_of(&sAr[buf][0]), xr + (size_t)mb * DIM_ + ka,
                TK, BM, DIM_, 3, 3);
    tdm_load_2d(lds_addr_of(&sAi[buf][0]), xi + (size_t)mb * DIM_ + ka,
                TK, BM, DIM_, 3, 3);
    tdm_load_2d(lds_addr_of(&sBr[buf][0]), wrt + ka * DIM_ + nb,
                BN, TK, DIM_, 4, 3);
    tdm_load_2d(lds_addr_of(&sBi[buf][0]), wit + ka * DIM_ + nb,
                BN, TK, DIM_, 4, 3);
  };

  v8f p1[2][2], p2[2][2], p3[2][2], p4[2][2];
#pragma unroll
  for (int i = 0; i < 2; ++i)
#pragma unroll
    for (int j = 0; j < 2; ++j) {
      p1[i][j] = zero8(); p2[i][j] = zero8();
      p3[i][j] = zero8(); p4[i][j] = zero8();
    }

  const int ksteps = DIM_ / TK;
  if (issuer) issue(0, 0);
  for (int kb = 0; kb < ksteps; ++kb) {
    const int cur = kb & 1;
    if (issuer) __builtin_amdgcn_s_wait_tensorcnt(0);  // buf[cur] landed
    __syncthreads();                                    // release all waves
    if (issuer && kb + 1 < ksteps) issue(kb + 1, cur ^ 1);

    uint4 aRlo[2], aRhi[2], aIlo[2], aIhi[2];
    load_a_frag(sAr[cur], wm + 0,  lane, aRlo[0], aRhi[0]);
    load_a_frag(sAr[cur], wm + 16, lane, aRlo[1], aRhi[1]);
    load_a_frag(sAi[cur], wm + 0,  lane, aIlo[0], aIhi[0]);
    load_a_frag(sAi[cur], wm + 16, lane, aIlo[1], aIhi[1]);
    uint4 bRlo[2], bRhi[2], bIlo[2], bIhi[2];
    load_b_frag(sBr[cur], wn + 0,  lane, bRlo[0], bRhi[0]);
    load_b_frag(sBr[cur], wn + 16, lane, bRlo[1], bRhi[1]);
    load_b_frag(sBi[cur], wn + 0,  lane, bIlo[0], bIhi[0]);
    load_b_frag(sBi[cur], wn + 16, lane, bIlo[1], bIhi[1]);

#pragma unroll
    for (int i = 0; i < 2; ++i)
#pragma unroll
      for (int j = 0; j < 2; ++j) {
        p1[i][j] = wmma_bf16(aRlo[i], aRhi[i], bRlo[j], bRhi[j], p1[i][j]);
        p2[i][j] = wmma_bf16(aIlo[i], aIhi[i], bIlo[j], bIhi[j], p2[i][j]);
        p3[i][j] = wmma_bf16(aIlo[i], aIhi[i], bRlo[j], bRhi[j], p3[i][j]);
        p4[i][j] = wmma_bf16(aRlo[i], aRhi[i], bIlo[j], bIhi[j], p4[i][j]);
      }
  }

  const int cn = lane & 15;
  const int cm = (lane >> 4) << 3;
#pragma unroll
  for (int i = 0; i < 2; ++i)
#pragma unroll
    for (int j = 0; j < 2; ++j) {
      const int m0 = mb + wm + i * 16 + cm;
      const int n0 = nb + wn + j * 16 + cn;
#pragma unroll
      for (int r = 0; r < 8; ++r) {
        yr[(size_t)(m0 + r) * DIM_ + n0] =
            (bf16_t)((p1[i][j][r] - p2[i][j][r]) * scale);
        yi[(size_t)(m0 + r) * DIM_ + n0] =
            (bf16_t)((p3[i][j][r] + p4[i][j][r]) * scale);
      }
    }
}

// ---------------------------------------------------------------------------
// Pass 2: causal scores = qr*kr^T + qi*ki^T (scale folded into q).
// K pre-transposed to [d][t] so B tiles are direct TDM loads.
// ---------------------------------------------------------------------------
__global__ __launch_bounds__(256) void scores_kernel(
    const bf16_t* __restrict__ q_r, const bf16_t* __restrict__ q_i,
    const bf16_t* __restrict__ k_rt, const bf16_t* __restrict__ k_it,
    bf16_t* __restrict__ scores) {
  const int tb = blockIdx.x * BN;
  const int sb = blockIdx.y * BM;
  if (tb >= sb + BM) return;        // fully above causal diagonal
  const int b = blockIdx.z;

  const bf16_t* qr  = q_r  + (size_t)b * S_ * DIM_;
  const bf16_t* qi  = q_i  + (size_t)b * S_ * DIM_;
  const bf16_t* krt = k_rt + (size_t)b * DIM_ * S_;
  const bf16_t* kit = k_it + (size_t)b * DIM_ * S_;
  bf16_t* sc = scores + (size_t)b * S_ * S_;

  const int tid  = threadIdx.x;
  const int lane = tid & 31;
  const int wave = tid >> 5;
  const int wm = (wave >> 1) << 5;
  const int wn = (wave & 1) << 5;
  const bool issuer = (__builtin_amdgcn_readfirstlane((int)threadIdx.x) == 0);

  __shared__ __align__(16) bf16_t sQr[2][BM * LDA];
  __shared__ __align__(16) bf16_t sQi[2][BM * LDA];
  __shared__ __align__(16) bf16_t sKr[2][TK * LDB];
  __shared__ __align__(16) bf16_t sKi[2][TK * LDB];

  auto issue = [&](int kb, int buf) {
    const size_t ka = (size_t)kb * TK;
    tdm_load_2d(lds_addr_of(&sQr[buf][0]), qr + (size_t)sb * DIM_ + ka,
                TK, BM, DIM_, 3, 3);
    tdm_load_2d(lds_addr_of(&sQi[buf][0]), qi + (size_t)sb * DIM_ + ka,
                TK, BM, DIM_, 3, 3);
    tdm_load_2d(lds_addr_of(&sKr[buf][0]), krt + ka * S_ + tb,
                BN, TK, S_, 4, 3);
    tdm_load_2d(lds_addr_of(&sKi[buf][0]), kit + ka * S_ + tb,
                BN, TK, S_, 4, 3);
  };

  v8f acc[2][2];
#pragma unroll
  for (int i = 0; i < 2; ++i)
#pragma unroll
    for (int j = 0; j < 2; ++j) acc[i][j] = zero8();

  const int ksteps = DIM_ / TK;
  if (issuer) issue(0, 0);
  for (int kb = 0; kb < ksteps; ++kb) {
    const int cur = kb & 1;
    if (issuer) __builtin_amdgcn_s_wait_tensorcnt(0);
    __syncthreads();
    if (issuer && kb + 1 < ksteps) issue(kb + 1, cur ^ 1);

    uint4 aRlo[2], aRhi[2], aIlo[2], aIhi[2];
    load_a_frag(sQr[cur], wm + 0,  lane, aRlo[0], aRhi[0]);
    load_a_frag(sQr[cur], wm + 16, lane, aRlo[1], aRhi[1]);
    load_a_frag(sQi[cur], wm + 0,  lane, aIlo[0], aIhi[0]);
    load_a_frag(sQi[cur], wm + 16, lane, aIlo[1], aIhi[1]);
    uint4 bRlo[2], bRhi[2], bIlo[2], bIhi[2];
    load_b_frag(sKr[cur], wn + 0,  lane, bRlo[0], bRhi[0]);
    load_b_frag(sKr[cur], wn + 16, lane, bRlo[1], bRhi[1]);
    load_b_frag(sKi[cur], wn + 0,  lane, bIlo[0], bIhi[0]);
    load_b_frag(sKi[cur], wn + 16, lane, bIlo[1], bIhi[1]);

#pragma unroll
    for (int i = 0; i < 2; ++i)
#pragma unroll
      for (int j = 0; j < 2; ++j) {
        acc[i][j] = wmma_bf16(aRlo[i], aRhi[i], bRlo[j], bRhi[j], acc[i][j]);
        acc[i][j] = wmma_bf16(aIlo[i], aIhi[i], bIlo[j], bIhi[j], acc[i][j]);
      }
  }

  const int cn = lane & 15;
  const int cm = (lane >> 4) << 3;
  const bf16_t NEG_INF = (bf16_t)(-__builtin_inff());
#pragma unroll
  for (int i = 0; i < 2; ++i)
#pragma unroll
    for (int j = 0; j < 2; ++j) {
      const int m0 = sb + wm + i * 16 + cm;
      const int n0 = tb + wn + j * 16 + cn;
#pragma unroll
      for (int r = 0; r < 8; ++r) {
        const int s = m0 + r;
        sc[(size_t)s * S_ + n0] = (n0 <= s) ? (bf16_t)acc[i][j][r] : NEG_INF;
      }
    }
}

// ---------------------------------------------------------------------------
// Pass 3: in-place row softmax over S=2048 (f32 math, bf16 storage).
// ---------------------------------------------------------------------------
__global__ __launch_bounds__(256) void softmax_kernel(bf16_t* __restrict__ scores) {
  const int row = blockIdx.x;        // b*S + s
  const int s   = row & (S_ - 1);
  bf16_t* p = scores + (size_t)row * S_;
  __shared__ float red[256];
  const int tid = threadIdx.x;

  float v[8];
  float mx = -3.0e38f;
#pragma unroll
  for (int j = 0; j < 8; ++j) {
    const int t = tid + j * 256;
    v[j] = (float)p[t];
    if (t <= s) mx = fmaxf(mx, v[j]);
  }
  red[tid] = mx;
  __syncthreads();
  for (int o = 128; o > 0; o >>= 1) {
    if (tid < o) red[tid] = fmaxf(red[tid], red[tid + o]);
    __syncthreads();
  }
  mx = red[0];
  __syncthreads();

  float se = 0.0f;
#pragma unroll
  for (int j = 0; j < 8; ++j) {
    const int t = tid + j * 256;
    const float e = (t <= s) ? __expf(v[j] - mx) : 0.0f;
    v[j] = e;
    se += e;
  }
  red[tid] = se;
  __syncthreads();
  for (int o = 128; o > 0; o >>= 1) {
    if (tid < o) red[tid] += red[tid + o];
    __syncthreads();
  }
  const float inv = 1.0f / red[0];
#pragma unroll
  for (int j = 0; j < 8; ++j) p[tid + j * 256] = (bf16_t)(v[j] * inv);
}

// ---------------------------------------------------------------------------
// Pass 4: out_r = attn @ vr, out_i = attn @ vi  (f32 output, causal K cut).
// ---------------------------------------------------------------------------
__global__ __launch_bounds__(256) void av_kernel(
    const bf16_t* __restrict__ attn, const bf16_t* __restrict__ v_r,
    const bf16_t* __restrict__ v_i, float* __restrict__ out) {
  const int nb = blockIdx.x * BN;
  const int sb = blockIdx.y * BM;
  const int b  = blockIdx.z;

  const bf16_t* at = attn + (size_t)b * S_ * S_;
  const bf16_t* vr = v_r + (size_t)b * S_ * DIM_;
  const bf16_t* vi = v_i + (size_t)b * S_ * DIM_;
  float* outR = out + (size_t)b * S_ * DIM_;
  float* outI = out + ((size_t)B_ + b) * S_ * DIM_;

  const int tid  = threadIdx.x;
  const int lane = tid & 31;
  const int wave = tid >> 5;
  const int wm = (wave >> 1) << 5;
  const int wn = (wave & 1) << 5;
  const bool issuer = (__builtin_amdgcn_readfirstlane((int)threadIdx.x) == 0);

  __shared__ __align__(16) bf16_t sA[2][BM * LDA];
  __shared__ __align__(16) bf16_t sVr[2][TK * LDB];
  __shared__ __align__(16) bf16_t sVi[2][TK * LDB];

  auto issue = [&](int kb, int buf) {
    const size_t ka = (size_t)kb * TK;
    tdm_load_2d(lds_addr_of(&sA[buf][0]), at + (size_t)sb * S_ + ka,
                TK, BM, S_, 3, 3);
    tdm_load_2d(lds_addr_of(&sVr[buf][0]), vr + ka * DIM_ + nb,
                BN, TK, DIM_, 4, 3);
    tdm_load_2d(lds_addr_of(&sVi[buf][0]), vi + ka * DIM_ + nb,
                BN, TK, DIM_, 4, 3);
  };

  v8f accR[2][2], accI[2][2];
#pragma unroll
  for (int i = 0; i < 2; ++i)
#pragma unroll
    for (int j = 0; j < 2; ++j) { accR[i][j] = zero8(); accI[i][j] = zero8(); }

  const int ksteps = (sb + BM) / TK;   // causal support: t <= s
  if (issuer) issue(0, 0);
  for (int kb = 0; kb < ksteps; ++kb) {
    const int cur = kb & 1;
    if (issuer) __builtin_amdgcn_s_wait_tensorcnt(0);
    __syncthreads();
    if (issuer && kb + 1 < ksteps) issue(kb + 1, cur ^ 1);

    uint4 aLo[2], aHi[2];
    load_a_frag(sA[cur], wm + 0,  lane, aLo[0], aHi[0]);
    load_a_frag(sA[cur], wm + 16, lane, aLo[1], aHi[1]);
    uint4 bRlo[2], bRhi[2], bIlo[2], bIhi[2];
    load_b_frag(sVr[cur], wn + 0,  lane, bRlo[0], bRhi[0]);
    load_b_frag(sVr[cur], wn + 16, lane, bRlo[1], bRhi[1]);
    load_b_frag(sVi[cur], wn + 0,  lane, bIlo[0], bIhi[0]);
    load_b_frag(sVi[cur], wn + 16, lane, bIlo[1], bIhi[1]);

#pragma unroll
    for (int i = 0; i < 2; ++i)
#pragma unroll
      for (int j = 0; j < 2; ++j) {
        accR[i][j] = wmma_bf16(aLo[i], aHi[i], bRlo[j], bRhi[j], accR[i][j]);
        accI[i][j] = wmma_bf16(aLo[i], aHi[i], bIlo[j], bIhi[j], accI[i][j]);
      }
  }

  const int cn = lane & 15;
  const int cm = (lane >> 4) << 3;
#pragma unroll
  for (int i = 0; i < 2; ++i)
#pragma unroll
    for (int j = 0; j < 2; ++j) {
      const int m0 = sb + wm + i * 16 + cm;
      const int n0 = nb + wn + j * 16 + cn;
#pragma unroll
      for (int r = 0; r < 8; ++r) {
        outR[(size_t)(m0 + r) * DIM_ + n0] = accR[i][j][r];
        outI[(size_t)(m0 + r) * DIM_ + n0] = accI[i][j][r];
      }
    }
}

// ---------------------------------------------------------------------------
// Host-side launch
// ---------------------------------------------------------------------------
extern "C" void kernel_launch(void* const* d_in, const int* in_sizes, int n_in,
                              void* d_out, int out_size, void* d_ws, size_t ws_size,
                              hipStream_t stream) {
  (void)in_sizes; (void)n_in; (void)out_size; (void)ws_size;
  const float* z_real = (const float*)d_in[0];
  const float* z_imag = (const float*)d_in[1];
  const float* wq_r = (const float*)d_in[2];
  const float* wq_i = (const float*)d_in[3];
  const float* wk_r = (const float*)d_in[4];
  const float* wk_i = (const float*)d_in[5];
  const float* wv_r = (const float*)d_in[6];
  const float* wv_i = (const float*)d_in[7];
  // d_in[8] (mask) is implemented analytically in-kernel.

  char* ws = (char*)d_ws;
  const size_t N_MD = (size_t)MTOT * DIM_;
  const size_t N_W  = (size_t)DIM_ * DIM_;
  const size_t SZ_MD = N_MD * sizeof(bf16_t);   // 16 MB
  const size_t SZ_W  = N_W * sizeof(bf16_t);    // 2 MB

  bf16_t* zr  = (bf16_t*)ws;
  bf16_t* zi  = (bf16_t*)(ws + SZ_MD);
  bf16_t* wt  = (bf16_t*)(ws + 2 * SZ_MD);      // 6 slots of DIM_*DIM_
  bf16_t* qr  = (bf16_t*)(ws + 2 * SZ_MD + 6 * SZ_W);
  bf16_t* qi  = qr + N_MD;
  bf16_t* kr  = qi + N_MD;
  bf16_t* ki  = kr + N_MD;
  bf16_t* vr  = ki + N_MD;
  bf16_t* vi  = vr + N_MD;
  bf16_t* sc  = vi + N_MD;                      // [B,S,S] bf16 scores/attn
  bf16_t* krt = sc + (size_t)B_ * S_ * S_;      // [B,D,S]
  bf16_t* kit = krt + (size_t)B_ * DIM_ * S_;

  const int nz = MTOT * DIM_;
  convert_kernel<<<nz / 1024, 256, 0, stream>>>(z_real, zr, nz);
  convert_kernel<<<nz / 1024, 256, 0, stream>>>(z_imag, zi, nz);

  const int wblocks = (DIM_ * DIM_) / 256;
  wtrans_kernel<<<wblocks, 256, 0, stream>>>(wq_r, wt + 0 * N_W);
  wtrans_kernel<<<wblocks, 256, 0, stream>>>(wq_i, wt + 1 * N_W);
  wtrans_kernel<<<wblocks, 256, 0, stream>>>(wk_r, wt + 2 * N_W);
  wtrans_kernel<<<wblocks, 256, 0, stream>>>(wk_i, wt + 3 * N_W);
  wtrans_kernel<<<wblocks, 256, 0, stream>>>(wv_r, wt + 4 * N_W);
  wtrans_kernel<<<wblocks, 256, 0, stream>>>(wv_i, wt + 5 * N_W);

  const float qscale = 0.03125f;  // 1024^-0.5, folded into Q
  dim3 pgrid(DIM_ / BN, MTOT / BM);
  proj_kernel<<<pgrid, 256, 0, stream>>>(zr, zi, wt + 0 * N_W, wt + 1 * N_W,
                                         qr, qi, qscale);
  proj_kernel<<<pgrid, 256, 0, stream>>>(zr, zi, wt + 2 * N_W, wt + 3 * N_W,
                                         kr, ki, 1.0f);
  proj_kernel<<<pgrid, 256, 0, stream>>>(zr, zi, wt + 4 * N_W, wt + 5 * N_W,
                                         vr, vi, 1.0f);

  const int tblocks = (int)(((size_t)B_ * DIM_ * S_) / 256);
  ktrans_kernel<<<tblocks, 256, 0, stream>>>(kr, krt);
  ktrans_kernel<<<tblocks, 256, 0, stream>>>(ki, kit);

  scores_kernel<<<dim3(S_ / BN, S_ / BM, B_), 256, 0, stream>>>(qr, qi, krt, kit, sc);
  softmax_kernel<<<B_ * S_, 256, 0, stream>>>(sc);
  av_kernel<<<dim3(DIM_ / BN, S_ / BM, B_), 256, 0, stream>>>(sc, vr, vi,
                                                              (float*)d_out);
}
